// CGCNN_6330781794983
// MI455X (gfx1250) — matrix operand order repaired and checked by reading, not compiled, not run on hardware
//
#include <hip/hip_runtime.h>
#include <math.h>

#define H_DIM    128
#define IN_D     92
#define EDGE_D   50
#define Z_DIM    306     // 2*H + EDGE_D
#define ZPAD     320     // K padded to 10 chunks of 32
#define POST_D   64
#define NLAYERS  4
#define BN_EPS   1e-5f
#define ASTRIDE  328     // LDS A-tile row stride (bf16 elems), 16B-aligned rows

typedef __attribute__((ext_vector_type(16))) __bf16 v16bf;
typedef __attribute__((ext_vector_type(8)))  float  v8f;

struct B32bits { int4 lo, hi; };

__device__ __forceinline__ v16bf make_v16bf(int4 lo, int4 hi) {
  B32bits t; t.lo = lo; t.hi = hi;
  return __builtin_bit_cast(v16bf, t);
}

__device__ __forceinline__ unsigned short f2bf(float f) {
  unsigned int u = __builtin_bit_cast(unsigned int, f);
  u += 0x7FFFu + ((u >> 16) & 1u);          // round-to-nearest-even
  return (unsigned short)(u >> 16);
}

// ---------------- weight pre-conversion: fp32 [L][306][128] -> bf16 [L][2][128][320] (transposed, K-padded)
extern "C" __global__ void convw_kernel(const float* __restrict__ Wf,
                                        const float* __restrict__ Ws,
                                        unsigned short* __restrict__ wt) {
  int idx = blockIdx.x * blockDim.x + threadIdx.x;
  const int total = NLAYERS * 2 * H_DIM * ZPAD;
  if (idx >= total) return;
  int kk  = idx % ZPAD;
  int c   = (idx / ZPAD) % H_DIM;
  int sel = (idx / (ZPAD * H_DIM)) % 2;
  int l   = idx / (ZPAD * H_DIM * 2);
  float v = 0.f;
  if (kk < Z_DIM) {
    const float* src = sel ? Ws : Wf;
    v = src[((size_t)l * Z_DIM + kk) * H_DIM + c];
  }
  wt[idx] = f2bf(v);
}

// ---------------- edge message kernel: gather -> bf16 WMMA (gate & softplus) -> atomic scatter
extern "C" __global__ __launch_bounds__(256)
void msg_kernel(const float* __restrict__ feat,
                const float* __restrict__ edge_attr,
                const unsigned short* __restrict__ WfT,
                const unsigned short* __restrict__ WsT,
                const float* __restrict__ bF, const float* __restrict__ bS,
                const int* __restrict__ ei, float* __restrict__ agg, int E) {
  __shared__ __align__(16) unsigned short zA[16 * ASTRIDE];
  __shared__ int sdst[16];
  __shared__ int ssrc[16];
  const int tid = threadIdx.x;
  const int e0  = blockIdx.x * 16;
  if (tid < 16) {
    int e = e0 + tid;
    sdst[tid] = (e < E) ? ei[E + e] : -1;   // edge_index[1] = dst
    ssrc[tid] = (e < E) ? ei[e]     : -1;   // edge_index[0] = src
  }
  __syncthreads();
  // build bf16 A tile: row r = edge-in-tile, cols = [feat[dst] | feat[src] | edge_attr | 0-pad]
  {
    const int r  = tid >> 4;
    const int lc = tid & 15;
    const int d = sdst[r], s = ssrc[r];
    const int e = e0 + r;
    for (int j = 0; j < 20; ++j) {
      int col = lc + j * 16;
      float v = 0.f;
      if (d >= 0) {
        if (col < H_DIM)          v = feat[(size_t)d * H_DIM + col];
        else if (col < 2 * H_DIM) v = feat[(size_t)s * H_DIM + (col - H_DIM)];
        else if (col < Z_DIM)     v = edge_attr[(size_t)e * EDGE_D + (col - 2 * H_DIM)];
      }
      zA[r * ASTRIDE + col] = f2bf(v);
    }
  }
  __syncthreads();

  const int wave = tid >> 5;        // 8 waves -> 8 x 16 output channels
  const int lane = tid & 31;
  const int half = lane >> 4;
  const int nm   = lane & 15;
  const int c    = wave * 16 + nm;

  v8f accF = {};
  v8f accS = {};
  const unsigned short* wfRow = WfT + (size_t)c * ZPAD;
  const unsigned short* wsRow = WsT + (size_t)c * ZPAD;
  for (int k = 0; k < 10; ++k) {
    // A (16x32 bf16): lane half -> K base, 16 contiguous K elems
    int kbA = k * 32 + half * 16;
    const int4* pa = (const int4*)(&zA[nm * ASTRIDE + kbA]);
    v16bf a = make_v16bf(pa[0], pa[1]);   // 16 contiguous bf16
    // B (32x16 bf16): lane half -> K base, 16 contiguous K elems
    int kbB = k * 32 + half * 16;
    const int4* pf = (const int4*)(wfRow + kbB);
    const int4* ps = (const int4*)(wsRow + kbB);
    v16bf bf16F = make_v16bf(pf[0], pf[1]);
    v16bf bf16S = make_v16bf(ps[0], ps[1]);
    accF = __builtin_amdgcn_wmma_f32_16x16x32_bf16(false, a, false, bf16F,
                                                   (short)0, accF, false, false);
    accS = __builtin_amdgcn_wmma_f32_16x16x32_bf16(false, a, false, bf16S,
                                                   (short)0, accS, false, false);
  }
  // epilogue: msg = sigmoid(f) * softplus(s), scatter-add to agg[dst]
  const float biasF = bF[c];
  const float biasS = bS[c];
  for (int rr = 0; rr < 8; ++rr) {
    int m = half * 8 + rr;          // C/D layout: M = vgpr + half*8
    int d = sdst[m];
    if (d >= 0) {
      float gt = 1.f / (1.f + expf(-(accF[rr] + biasF)));
      float xs = accS[rr] + biasS;
      float sp = (xs > 20.f) ? xs : log1pf(expf(xs));
      atomicAdd(&agg[(size_t)d * H_DIM + c], gt * sp);
    }
  }
}

// ---------------- pre FC: h = x @ pre_W + pre_b
extern "C" __global__ __launch_bounds__(128)
void prefc_kernel(const float* __restrict__ x, const float* __restrict__ W,
                  const float* __restrict__ b, float* __restrict__ h, int N) {
  __shared__ float xs[IN_D];
  int n = blockIdx.x;
  int c = threadIdx.x;
  if (c < IN_D) xs[c] = x[(size_t)n * IN_D + c];
  __syncthreads();
  float acc = b[c];
  for (int k = 0; k < IN_D; ++k) acc += xs[k] * W[k * H_DIM + c];
  h[(size_t)n * H_DIM + c] = acc;
}

// ---------------- residual add + per-channel sum/sumsq (h may alias badd)
extern "C" __global__ __launch_bounds__(128)
void stats_kernel(const float* __restrict__ a, const float* __restrict__ badd,
                  float* __restrict__ h, float* __restrict__ sum,
                  float* __restrict__ sumsq, int N, int rows) {
  int c  = threadIdx.x;
  int n0 = blockIdx.x * rows;
  int n1 = min(n0 + rows, N);
  float s = 0.f, s2 = 0.f;
  for (int n = n0; n < n1; ++n) {
    size_t i = (size_t)n * H_DIM + c;
    float v = a[i];
    if (badd) v += badd[i];
    h[i] = v;
    s += v; s2 += v * v;
  }
  atomicAdd(&sum[c], s);
  atomicAdd(&sumsq[c], s2);
}

// ---------------- BN (biased var) + ReLU
extern "C" __global__ void bnrelu_kernel(const float* __restrict__ h, float* __restrict__ out,
                                         const float* __restrict__ sum, const float* __restrict__ sumsq,
                                         const float* __restrict__ g, const float* __restrict__ b, int N) {
  size_t i = (size_t)blockIdx.x * blockDim.x + threadIdx.x;
  if (i >= (size_t)N * H_DIM) return;
  int c = (int)(i & (H_DIM - 1));
  float invN = 1.f / (float)N;
  float mu  = sum[c] * invN;
  float var = sumsq[c] * invN - mu * mu;
  float v = g[c] * (h[i] - mu) * rsqrtf(var + BN_EPS) + b[c];
  out[i] = fmaxf(v, 0.f);
}

// ---------------- global mean pool (sums + counts)
extern "C" __global__ __launch_bounds__(128)
void pool_kernel(const float* __restrict__ feat, const int* __restrict__ batch,
                 float* __restrict__ pooled, float* __restrict__ cnt, int N, int rows) {
  int c  = threadIdx.x;
  int n0 = blockIdx.x * rows;
  int n1 = min(n0 + rows, N);
  for (int n = n0; n < n1; ++n) {
    int gi = batch[n];
    atomicAdd(&pooled[(size_t)gi * H_DIM + c], feat[(size_t)n * H_DIM + c]);
    if (c == 0) atomicAdd(&cnt[gi], 1.f);
  }
}

// ---------------- post FC on pooled means + stats for post-BN
extern "C" __global__ __launch_bounds__(64)
void post_kernel(const float* __restrict__ pooled, const float* __restrict__ cnt,
                 const float* __restrict__ W, const float* __restrict__ b,
                 float* __restrict__ hpost, float* __restrict__ psum,
                 float* __restrict__ psumsq) {
  __shared__ float p[H_DIM];
  int gi = blockIdx.x;
  int c  = threadIdx.x;
  float inv = 1.f / fmaxf(cnt[gi], 1.f);
  p[c]      = pooled[(size_t)gi * H_DIM + c] * inv;
  p[c + 64] = pooled[(size_t)gi * H_DIM + c + 64] * inv;
  __syncthreads();
  float acc = b[c];
  for (int k = 0; k < H_DIM; ++k) acc += p[k] * W[k * POST_D + c];
  hpost[(size_t)gi * POST_D + c] = acc;
  atomicAdd(&psum[c], acc);
  atomicAdd(&psumsq[c], acc * acc);
}

// ---------------- post BN + ReLU + output head
extern "C" __global__ void final_kernel(const float* __restrict__ hpost,
                                        const float* __restrict__ psum, const float* __restrict__ psumsq,
                                        const float* __restrict__ g, const float* __restrict__ b,
                                        const float* __restrict__ outW, const float* __restrict__ outb,
                                        float* __restrict__ out, int G) {
  int gi = blockIdx.x * blockDim.x + threadIdx.x;
  if (gi >= G) return;
  float invG = 1.f / (float)G;
  float acc = outb[0];
  for (int c = 0; c < POST_D; ++c) {
    float mu  = psum[c] * invG;
    float var = psumsq[c] * invG - mu * mu;
    float h = hpost[(size_t)gi * POST_D + c];
    float v = g[c] * (h - mu) * rsqrtf(var + BN_EPS) + b[c];
    acc += fmaxf(v, 0.f) * outW[c];
  }
  out[gi] = acc;
}

extern "C" void kernel_launch(void* const* d_in, const int* in_sizes, int n_in,
                              void* d_out, int out_size, void* d_ws, size_t ws_size,
                              hipStream_t stream) {
  const float* x         = (const float*)d_in[0];
  const float* edge_attr = (const float*)d_in[1];
  const float* pre_W     = (const float*)d_in[2];
  const float* pre_b     = (const float*)d_in[3];
  const float* pre_bn_g  = (const float*)d_in[4];
  const float* pre_bn_b  = (const float*)d_in[5];
  const float* conv_Wf   = (const float*)d_in[6];
  const float* conv_bf   = (const float*)d_in[7];
  const float* conv_Ws   = (const float*)d_in[8];
  const float* conv_bs   = (const float*)d_in[9];
  const float* bn_g      = (const float*)d_in[10];
  const float* bn_b      = (const float*)d_in[11];
  const float* post_W    = (const float*)d_in[12];
  const float* post_b    = (const float*)d_in[13];
  const float* post_bn_g = (const float*)d_in[14];
  const float* post_bn_b = (const float*)d_in[15];
  const float* out_W     = (const float*)d_in[16];
  const float* out_b     = (const float*)d_in[17];
  const int*   ei        = (const int*)d_in[18];
  const int*   batch     = (const int*)d_in[19];
  float* out = (float*)d_out;

  const int N = in_sizes[0] / IN_D;
  const int E = in_sizes[1] / EDGE_D;
  const int G = out_size;

  // workspace layout
  float* feat = (float*)d_ws;                                   // N*H
  float* agg  = feat + (size_t)N * H_DIM;                       // N*H (also h buffer)
  unsigned short* wt = (unsigned short*)(agg + (size_t)N * H_DIM); // L*2*H*ZPAD bf16
  float* sums   = (float*)(wt + (size_t)NLAYERS * 2 * H_DIM * ZPAD);
  float* sumsq  = sums + H_DIM;
  float* pooled = sumsq + H_DIM;                                // G*H
  float* cnt    = pooled + (size_t)G * H_DIM;                   // G
  float* hpost  = cnt + G;                                      // G*POST
  float* psum   = hpost + (size_t)G * POST_D;                   // POST
  float* psumsq = psum + POST_D;                                // POST

  // 1) convert conv weights to transposed bf16 (K padded to 320)
  {
    int total = NLAYERS * 2 * H_DIM * ZPAD;
    convw_kernel<<<(total + 255) / 256, 256, 0, stream>>>(conv_Wf, conv_Ws, wt);
  }

  // 2) pre FC -> BN -> ReLU into feat
  prefc_kernel<<<N, H_DIM, 0, stream>>>(x, pre_W, pre_b, agg, N);
  hipMemsetAsync(sums, 0, 2 * H_DIM * sizeof(float), stream);
  const int ROWS = 128;
  const int sblocks = (N + ROWS - 1) / ROWS;
  stats_kernel<<<sblocks, H_DIM, 0, stream>>>(agg, nullptr, agg, sums, sumsq, N, ROWS);
  const size_t tot = (size_t)N * H_DIM;
  const int bnblocks = (int)((tot + 255) / 256);
  bnrelu_kernel<<<bnblocks, 256, 0, stream>>>(agg, feat, sums, sumsq, pre_bn_g, pre_bn_b, N);

  // 3) CGConv layers
  const int etiles = (E + 15) / 16;
  for (int l = 0; l < NLAYERS; ++l) {
    hipMemsetAsync(agg, 0, (size_t)N * H_DIM * sizeof(float), stream);
    const unsigned short* WfT = wt + (size_t)(l * 2 + 0) * H_DIM * ZPAD;
    const unsigned short* WsT = wt + (size_t)(l * 2 + 1) * H_DIM * ZPAD;
    msg_kernel<<<etiles, 256, 0, stream>>>(feat, edge_attr, WfT, WsT,
                                           conv_bf + l * H_DIM, conv_bs + l * H_DIM,
                                           ei, agg, E);
    hipMemsetAsync(sums, 0, 2 * H_DIM * sizeof(float), stream);
    stats_kernel<<<sblocks, H_DIM, 0, stream>>>(feat, agg, agg, sums, sumsq, N, ROWS);
    bnrelu_kernel<<<bnblocks, 256, 0, stream>>>(agg, feat, sums, sumsq,
                                                bn_g + l * H_DIM, bn_b + l * H_DIM, N);
  }

  // 4) global mean pool
  hipMemsetAsync(pooled, 0, ((size_t)G * H_DIM + G) * sizeof(float), stream);
  pool_kernel<<<sblocks, H_DIM, 0, stream>>>(feat, batch, pooled, cnt, N, ROWS);

  // 5) post FC + BN + ReLU + head
  hipMemsetAsync(psum, 0, 2 * POST_D * sizeof(float), stream);
  post_kernel<<<G, POST_D, 0, stream>>>(pooled, cnt, post_W, post_b, hpost, psum, psumsq);
  final_kernel<<<(G + 255) / 256, 256, 0, stream>>>(hpost, psum, psumsq,
                                                    post_bn_g, post_bn_b, out_W, out_b, out, G);
}
